// CAL_GAT_19550691131407
// MI455X (gfx1250) — matrix operand order repaired and checked by reading, not compiled, not run on hardware
//
#include <hip/hip_runtime.h>
#include <hip/hip_bf16.h>

typedef __attribute__((ext_vector_type(16))) __bf16 v16bf;
typedef __attribute__((ext_vector_type(8)))  __bf16 v8bf;
typedef __attribute__((ext_vector_type(8)))  float  v8f;

#define N_NODES 50000
#define N_EDGES 800000
#define FIN     128
#define HIDF    32
#define HEADS   8
#define HDF     256
#define NCLS    10
#define NGRAPH  128
#define EPSBN   1e-5f
#define SLOPE   0.2f

static __device__ __forceinline__ float leaky(float v) {
    return v > 0.f ? v : v * SLOPE;
}

static __device__ __forceinline__ void atomicMaxF(float* addr, float val) {
    unsigned* ua = (unsigned*)addr;
    unsigned old = *ua;
    while (__uint_as_float(old) < val) {
        unsigned assumed = old;
        old = atomicCAS(ua, assumed, __float_as_uint(val));
        if (old == assumed) break;
    }
}

// ---------------------------------------------------------------------------
// Generic utility kernels
// ---------------------------------------------------------------------------
__global__ void fill_kernel(float* p, float v, int n) {
    int i = blockIdx.x * blockDim.x + threadIdx.x;
    if (i < n) p[i] = v;
}

__global__ void act_kernel(float* x, int n, int act) {  // 1=relu, 2=elu
    int i = blockIdx.x * blockDim.x + threadIdx.x;
    if (i >= n) return;
    float v = x[i];
    if (act == 1) v = fmaxf(v, 0.f);
    else if (act == 2) v = v > 0.f ? v : (__expf(v) - 1.f);
    x[i] = v;
}

__global__ void add_bias_kernel(float* x, const float* b, int rows, int C) {
    int i = blockIdx.x * blockDim.x + threadIdx.x;
    if (i >= rows * C) return;
    x[i] += b[i % C];
}

// ---------------------------------------------------------------------------
// BatchNorm (training-mode, biased variance)
// ---------------------------------------------------------------------------
__global__ void bn_stats_kernel(const float* __restrict__ x, int rows, int C,
                                float* mean, float* rstd) {
    int c = blockIdx.x;
    float s = 0.f, s2 = 0.f;
    for (int r = threadIdx.x; r < rows; r += blockDim.x) {
        float v = x[(size_t)r * C + c];
        s += v; s2 += v * v;
    }
    __shared__ float ss[256], ss2[256];
    ss[threadIdx.x] = s; ss2[threadIdx.x] = s2;
    __syncthreads();
    for (int st = 128; st; st >>= 1) {
        if ((int)threadIdx.x < st) {
            ss[threadIdx.x] += ss[threadIdx.x + st];
            ss2[threadIdx.x] += ss2[threadIdx.x + st];
        }
        __syncthreads();
    }
    if (threadIdx.x == 0) {
        float m = ss[0] / rows;
        float v = ss2[0] / rows - m * m;
        mean[c] = m;
        rstd[c] = rsqrtf(v + EPSBN);
    }
}

__global__ void bn_apply_kernel(float* x, const float* mean, const float* rstd,
                                const float* gam, const float* bet,
                                int rows, int C, int act) {
    int i = blockIdx.x * blockDim.x + threadIdx.x;
    if (i >= rows * C) return;
    int c = i % C;
    float v = (x[i] - mean[c]) * rstd[c] * gam[c] + bet[c];
    if (act == 1) v = fmaxf(v, 0.f);
    else if (act == 2) v = v > 0.f ? v : (__expf(v) - 1.f);
    x[i] = v;
}

// ---------------------------------------------------------------------------
// bf16 packing for WMMA GEMM
// ---------------------------------------------------------------------------
// A: straight row-major f32 -> bf16 copy
__global__ void pack_a_kernel(const float* __restrict__ A, __bf16* __restrict__ Ap,
                              long long n) {
    long long i = (long long)blockIdx.x * blockDim.x + threadIdx.x;
    if (i < n) Ap[i] = (__bf16)A[i];
}

// B[K,Nn] row-major f32 -> WMMA B-fragment layout:
//   Bp[((kchunk*tiles_n + nt)*32 + lane)*16 + j], lanes 0-15 K={0..7,16..23}; lanes 16-31 K={8..15,24..31}
__global__ void pack_b_kernel(const float* __restrict__ B, __bf16* __restrict__ Bp,
                              int K, int Nn) {
    int tiles_n = Nn >> 4;
    int total = (K >> 5) * tiles_n * 32;
    int i = blockIdx.x * blockDim.x + threadIdx.x;
    if (i >= total) return;
    int lane = i & 31;
    int rest = i >> 5;
    int nt = rest % tiles_n, kc = rest / tiles_n;
    int half = lane >> 4, l16 = lane & 15;
    int n = nt * 16 + l16;
    __bf16* op = Bp + (size_t)i * 16;
#pragma unroll
    for (int j = 0; j < 16; ++j)
        op[j] = (__bf16)B[(size_t)(kc * 32 + (j < 8 ? 8 * half + j : 16 + 8 * half + (j - 8))) * Nn + n];
}

// ---------------------------------------------------------------------------
// bf16 WMMA GEMM: C[M,Nn] = A[M,K] @ B[K,Nn]; one 16x(16*NT) strip per wave.
// A fragment loaded once per K-chunk, reused across NT wmma ops.
// Requires M%16==0, Nn%(16*NT)==0, K%32==0.
// ---------------------------------------------------------------------------
template <int NT>
__global__ void __launch_bounds__(256)
gemm_wmma_bf16(const __bf16* __restrict__ Ap, const __bf16* __restrict__ Bp,
               float* __restrict__ C, int M, int Nn, int K) {
    const int lane = threadIdx.x & 31;
    const int wave = threadIdx.x >> 5;
    const int tiles_n = Nn >> 4;
    const int strips_n = tiles_n / NT;
    const int total = (M >> 4) * strips_n;
    const int strip = blockIdx.x * (blockDim.x >> 5) + wave;
    if (strip >= total) return;  // whole-wave uniform: EXEC stays all-ones
    const int mt = strip / strips_n, ns = strip % strips_n;
    const int half = lane >> 4;
    const int l16  = lane & 15;
    const int arow = mt * 16 + l16;  // A: M = lane%16

    v8f acc[NT];
#pragma unroll
    for (int t = 0; t < NT; ++t) acc[t] = (v8f){};

    const __bf16* arowp = Ap + (size_t)arow * K;
    for (int kk = 0; kk < K; kk += 32) {
        // A 16x32 bf16: lanes 0-15 hold K={0..7,16..23}; lanes 16-31 K={8..15,24..31}
        v8bf lo = *(const v8bf*)(arowp + kk + 8 * half);
        v8bf hi = *(const v8bf*)(arowp + kk + 16 + 8 * half);
        v16bf a;
#pragma unroll
        for (int i = 0; i < 8; ++i) { a[i] = lo[i]; a[i + 8] = hi[i]; }
        const int kc = kk >> 5;
        const __bf16* bbase = Bp + ((size_t)kc * tiles_n + ns * NT) * 32 * 16;
#pragma unroll
        for (int t = 0; t < NT; ++t) {
            const v16bf b = *(const v16bf*)(bbase + ((size_t)t * 32 + lane) * 16);
            acc[t] = __builtin_amdgcn_wmma_f32_16x16x32_bf16(
                false, a, false, b, (short)0, acc[t], false, false);
        }
    }
    // C/D: VGPR v -> M = v + 8*half, N = lane%16
#pragma unroll
    for (int t = 0; t < NT; ++t) {
        const int ncol = (ns * NT + t) * 16 + l16;
#pragma unroll
        for (int v = 0; v < 8; ++v) {
            C[(size_t)(mt * 16 + v + 8 * half) * Nn + ncol] = acc[t][v];
        }
    }
}

// ---------------------------------------------------------------------------
// GCN: degree / norm / scatter
// ---------------------------------------------------------------------------
__global__ void deg_kernel(const int* __restrict__ col, const float* __restrict__ ew,
                           int ewstride, int ewoff, float* deg, int E) {
    int e = blockIdx.x * blockDim.x + threadIdx.x;
    if (e >= E) return;
    float w = ew ? ew[(size_t)e * ewstride + ewoff] : 1.f;
    atomicAdd(&deg[col[e]], w);
}

__global__ void rsqrt_kernel(float* d, int n) {
    int i = blockIdx.x * blockDim.x + threadIdx.x;
    if (i < n) d[i] = d[i] > 0.f ? rsqrtf(d[i]) : 0.f;
}

// node-level self-loop init: out[n] = bias + tmp[n]*dinv[n]^2
__global__ void gcn_self_node(const float* __restrict__ tmp, const float* __restrict__ dinv,
                              const float* __restrict__ bias, float* out, int Nn, int C) {
    int i = blockIdx.x * blockDim.x + threadIdx.x;
    if (i >= Nn * C) return;
    int n = i / C, f = i % C;
    float di = dinv[n];
    out[i] = bias[f] + tmp[i] * di * di;
}

// edge scatter: out[idx(col)] += tmp[row]*dinv[row]*w*dinv[col], 4 feats/thread
__global__ void gcn_scatter_kernel(const float* __restrict__ tmp,
                                   const int* __restrict__ rowi, const int* __restrict__ coli,
                                   const float* __restrict__ dinv,
                                   const float* __restrict__ ew, int ewstride, int ewoff,
                                   const int* __restrict__ batchmap,
                                   float* out, int E, int C) {
    int chunks = C >> 2;
    long long i = (long long)blockIdx.x * blockDim.x + threadIdx.x;
    if (i >= (long long)E * chunks) return;
    int e = (int)(i / chunks), q = (int)(i % chunks), f = q * 4;
    int r = rowi[e], c = coli[e];
    float w = ew ? ew[(size_t)e * ewstride + ewoff] : 1.f;
    float nrm = dinv[r] * w * dinv[c];
    const float4 v = *(const float4*)(tmp + (size_t)r * C + f);
    int o = batchmap ? batchmap[c] : c;
    float* op = out + (size_t)o * C + f;
    atomicAdd(op + 0, v.x * nrm);
    atomicAdd(op + 1, v.y * nrm);
    atomicAdd(op + 2, v.z * nrm);
    atomicAdd(op + 3, v.w * nrm);
}

// pooled self-loop: pc[batch[n]] += tmp[n]*dinv[n]^2
__global__ void gcn_self_pool(const float* __restrict__ tmp, const float* __restrict__ dinv,
                              const int* __restrict__ batch, float* pc, int Nn, int C) {
    int chunks = C >> 2;
    long long i = (long long)blockIdx.x * blockDim.x + threadIdx.x;
    if (i >= (long long)Nn * chunks) return;
    int n = (int)(i / chunks), f = (int)(i % chunks) * 4;
    float di = dinv[n]; float nrm = di * di;
    const float4 v = *(const float4*)(tmp + (size_t)n * C + f);
    float* op = pc + (size_t)batch[n] * C + f;
    atomicAdd(op + 0, v.x * nrm);
    atomicAdd(op + 1, v.y * nrm);
    atomicAdd(op + 2, v.z * nrm);
    atomicAdd(op + 3, v.w * nrm);
}

// ---------------------------------------------------------------------------
// GAT attention pipeline
// ---------------------------------------------------------------------------
__global__ void att_score_kernel(const float* __restrict__ hW,
                                 const float* __restrict__ a_s, const float* __restrict__ a_d,
                                 float* asrc, float* adst, int Nn) {
    int i = blockIdx.x * blockDim.x + threadIdx.x;
    if (i >= Nn * HEADS) return;
    int n = i >> 3, hh = i & 7;
    const float* p  = hW + (size_t)n * HDF + hh * HIDF;
    const float* as_ = a_s + hh * HIDF;
    const float* ad_ = a_d + hh * HIDF;
    float s = 0.f, d = 0.f;
#pragma unroll 8
    for (int j = 0; j < HIDF; ++j) { float v = p[j]; s += v * as_[j]; d += v * ad_[j]; }
    asrc[i] = s; adst[i] = d;
}

__global__ void att_init_self(const float* asrc, const float* adst, float* emax, int Nn) {
    int i = blockIdx.x * blockDim.x + threadIdx.x;
    if (i >= Nn * HEADS) return;
    emax[i] = leaky(asrc[i] + adst[i]);
}

__global__ void att_max_edges(const float* __restrict__ asrc, const float* __restrict__ adst,
                              const int* __restrict__ rowi, const int* __restrict__ coli,
                              float* emax, int E) {
    int i = blockIdx.x * blockDim.x + threadIdx.x;
    if (i >= E * HEADS) return;
    int e = i >> 3, hh = i & 7;
    float v = leaky(asrc[rowi[e] * HEADS + hh] + adst[coli[e] * HEADS + hh]);
    atomicMaxF(&emax[coli[e] * HEADS + hh], v);
}

__global__ void att_den_init(const float* asrc, const float* adst, const float* emax,
                             float* den, int Nn) {
    int i = blockIdx.x * blockDim.x + threadIdx.x;
    if (i >= Nn * HEADS) return;
    den[i] = __expf(leaky(asrc[i] + adst[i]) - emax[i]);
}

__global__ void att_den_edges(const float* __restrict__ asrc, const float* __restrict__ adst,
                              const int* __restrict__ rowi, const int* __restrict__ coli,
                              const float* __restrict__ emax, float* den, int E) {
    int i = blockIdx.x * blockDim.x + threadIdx.x;
    if (i >= E * HEADS) return;
    int e = i >> 3, hh = i & 7;
    int c = coli[e];
    float v = leaky(asrc[rowi[e] * HEADS + hh] + adst[c * HEADS + hh]);
    atomicAdd(&den[c * HEADS + hh], __expf(v - emax[c * HEADS + hh]));
}

__global__ void att_alpha_edges(const float* __restrict__ asrc, const float* __restrict__ adst,
                                const int* __restrict__ rowi, const int* __restrict__ coli,
                                const float* __restrict__ emax, const float* __restrict__ den,
                                float* ealpha, int E) {
    int i = blockIdx.x * blockDim.x + threadIdx.x;
    if (i >= E * HEADS) return;
    int e = i >> 3, hh = i & 7;
    int c = coli[e];
    float v = leaky(asrc[rowi[e] * HEADS + hh] + adst[c * HEADS + hh]);
    ealpha[i] = __expf(v - emax[c * HEADS + hh]) / den[c * HEADS + hh];
}

// out[n] = bias + hW[n] * alpha_self   (self-loop contribution, initializes out)
__global__ void att_out_init(const float* __restrict__ hW,
                             const float* __restrict__ asrc, const float* __restrict__ adst,
                             const float* __restrict__ emax, const float* __restrict__ den,
                             const float* __restrict__ bias, float* out, int Nn) {
    int i = blockIdx.x * blockDim.x + threadIdx.x;
    if (i >= Nn * HDF) return;
    int n = i / HDF, f = i % HDF, hh = f >> 5;
    int a = n * HEADS + hh;
    float alpha = __expf(leaky(asrc[a] + adst[a]) - emax[a]) / den[a];
    out[i] = bias[f] + hW[i] * alpha;
}

// out[col] += hW[row] * alpha[e,head],  4 feats/thread
__global__ void att_scatter(const float* __restrict__ hW,
                            const int* __restrict__ rowi, const int* __restrict__ coli,
                            const float* __restrict__ ealpha, float* out, int E) {
    long long i = (long long)blockIdx.x * blockDim.x + threadIdx.x;
    if (i >= (long long)E * (HDF / 4)) return;
    int e = (int)(i / (HDF / 4)), q = (int)(i % (HDF / 4));
    int f = q * 4, hh = q >> 3;
    float alpha = ealpha[e * HEADS + hh];
    const float4 v = *(const float4*)(hW + (size_t)rowi[e] * HDF + f);
    float* op = out + (size_t)coli[e] * HDF + f;
    atomicAdd(op + 0, v.x * alpha);
    atomicAdd(op + 1, v.y * alpha);
    atomicAdd(op + 2, v.z * alpha);
    atomicAdd(op + 3, v.w * alpha);
}

// ---------------------------------------------------------------------------
// Edge attention scores: one wave per edge, softmax over 2 outputs
// ---------------------------------------------------------------------------
__global__ void edge_score_kernel(const float* __restrict__ h,
                                  const int* __restrict__ rowi, const int* __restrict__ coli,
                                  const float* __restrict__ Wea, const float* __restrict__ bea,
                                  float* escore, int E) {
    int wv = blockIdx.x * (blockDim.x >> 5) + (threadIdx.x >> 5);
    if (wv >= E) return;
    int lane = threadIdx.x & 31;
    int r = rowi[wv], c = coli[wv];
    float s0 = 0.f, s1 = 0.f;
    for (int j = lane; j < HDF; j += 32) {
        float hr = h[(size_t)r * HDF + j];
        float hc = h[(size_t)c * HDF + j];
        s0 += hr * Wea[j * 2 + 0] + hc * Wea[(HDF + j) * 2 + 0];
        s1 += hr * Wea[j * 2 + 1] + hc * Wea[(HDF + j) * 2 + 1];
    }
    for (int off = 16; off; off >>= 1) {
        s0 += __shfl_xor(s0, off, 32);
        s1 += __shfl_xor(s1, off, 32);
    }
    if (lane == 0) {
        s0 += bea[0]; s1 += bea[1];
        float m = fmaxf(s0, s1);
        float e0 = __expf(s0 - m), e1 = __expf(s1 - m);
        float inv = 1.f / (e0 + e1);
        escore[(size_t)wv * 2 + 0] = e0 * inv;
        escore[(size_t)wv * 2 + 1] = e1 * inv;
    }
}

// ---------------------------------------------------------------------------
// Pooling / classifier tail
// ---------------------------------------------------------------------------
__global__ void pool_count(const int* batch, float* cnt, int Nn) {
    int i = blockIdx.x * blockDim.x + threadIdx.x;
    if (i < Nn) atomicAdd(&cnt[batch[i]], 1.f);
}

__global__ void pool_finalize(float* pc, const float* cnt, const float* bias, int Gn, int C) {
    int i = blockIdx.x * blockDim.x + threadIdx.x;
    if (i >= Gn * C) return;
    int g = i / C, f = i % C;
    pc[i] = pc[i] / fmaxf(cnt[g], 1.f) + bias[f];
}

__global__ void concat_kernel(const float* pc, const float* pt, float* comb, int Gn) {
    int i = blockIdx.x * blockDim.x + threadIdx.x;
    if (i >= Gn * 2 * HDF) return;
    int g = i / (2 * HDF), f = i % (2 * HDF);
    comb[i] = (f < HDF) ? pc[g * HDF + f] : pt[g * HDF + (f - HDF)];
}

__global__ void final_linear_kernel(const float* __restrict__ hcls, const float* __restrict__ W2,
                                    const float* __restrict__ b2, float* logits, int Gn) {
    int i = blockIdx.x * blockDim.x + threadIdx.x;
    if (i >= Gn * NCLS) return;
    int g = i / NCLS, c = i % NCLS;
    float s = b2[c];
#pragma unroll 8
    for (int j = 0; j < HIDF; ++j) s += hcls[g * HIDF + j] * W2[j * NCLS + c];
    logits[i] = s;
}

__global__ void log_softmax_kernel(const float* logits, float* out, int Gn) {
    int g = blockIdx.x * blockDim.x + threadIdx.x;
    if (g >= Gn) return;
    float m = -1e30f;
    for (int c = 0; c < NCLS; ++c) m = fmaxf(m, logits[g * NCLS + c]);
    float s = 0.f;
    for (int c = 0; c < NCLS; ++c) s += __expf(logits[g * NCLS + c] - m);
    float lse = m + __logf(s);
    for (int c = 0; c < NCLS; ++c) out[g * NCLS + c] = logits[g * NCLS + c] - lse;
}

// ---------------------------------------------------------------------------
// Host-side orchestration
// ---------------------------------------------------------------------------
static inline int cdiv(long long a, long long b) { return (int)((a + b - 1) / b); }

extern "C" void kernel_launch(void* const* d_in, const int* in_sizes, int n_in,
                              void* d_out, int out_size, void* d_ws, size_t ws_size,
                              hipStream_t stream) {
    const float* x       = (const float*)d_in[0];
    const int*   eidx    = (const int*)d_in[1];
    const int*   batch   = (const int*)d_in[2];
    const float* W_init  = (const float*)d_in[3];
    const float* b_init  = (const float*)d_in[4];
    const float* gam0    = (const float*)d_in[5];
    const float* bet0    = (const float*)d_in[6];
    const float* W1      = (const float*)d_in[7];
    const float* as1     = (const float*)d_in[8];
    const float* ad1     = (const float*)d_in[9];
    const float* b1      = (const float*)d_in[10];
    const float* gam1    = (const float*)d_in[11];
    const float* bet1    = (const float*)d_in[12];
    const float* W2      = (const float*)d_in[13];
    const float* as2     = (const float*)d_in[14];
    const float* ad2     = (const float*)d_in[15];
    const float* b2      = (const float*)d_in[16];
    const float* gam2    = (const float*)d_in[17];
    const float* bet2    = (const float*)d_in[18];
    const float* Wea     = (const float*)d_in[19];
    const float* bea     = (const float*)d_in[20];
    const float* Wgc     = (const float*)d_in[21];
    const float* bgc     = (const float*)d_in[22];
    const float* Wgt     = (const float*)d_in[23];
    const float* bgt     = (const float*)d_in[24];
    const float* clsW1[3] = { (const float*)d_in[25], (const float*)d_in[31], (const float*)d_in[37] };
    const float* clsB1[3] = { (const float*)d_in[26], (const float*)d_in[32], (const float*)d_in[38] };
    const float* clsG [3] = { (const float*)d_in[27], (const float*)d_in[33], (const float*)d_in[39] };
    const float* clsBt[3] = { (const float*)d_in[28], (const float*)d_in[34], (const float*)d_in[40] };
    const float* clsW2[3] = { (const float*)d_in[29], (const float*)d_in[35], (const float*)d_in[41] };
    const float* clsB2[3] = { (const float*)d_in[30], (const float*)d_in[36], (const float*)d_in[42] };

    const int* rowi = eidx;
    const int* coli = eidx + N_EDGES;

    // -------- workspace carve-up (floats; all chunk sizes multiples of 8) ----
    float* ws = (float*)d_ws;
    size_t o = 0;
    auto alloc = [&](size_t n) { float* p = ws + o; o += n; return p; };
    float* deg    = alloc(N_NODES);
    float* dinv   = alloc(N_NODES);
    float* h0     = alloc((size_t)N_NODES * HIDF);
    float* tmp    = alloc((size_t)N_NODES * HDF);
    float* hA     = alloc((size_t)N_NODES * HDF);
    float* hB     = alloc((size_t)N_NODES * HDF);
    float* asrc   = alloc((size_t)N_NODES * HEADS);
    float* adst   = alloc((size_t)N_NODES * HEADS);
    float* emax   = alloc((size_t)N_NODES * HEADS);
    float* den    = alloc((size_t)N_NODES * HEADS);
    float* ealpha = alloc((size_t)N_EDGES * HEADS);
    float* escore = alloc((size_t)N_EDGES * 2);
    float* pc     = alloc((size_t)NGRAPH * HDF);
    float* pt     = alloc((size_t)NGRAPH * HDF);
    float* cnt    = alloc(NGRAPH);
    float* comb   = alloc((size_t)NGRAPH * 2 * HDF);
    float* clsh   = alloc((size_t)NGRAPH * HIDF);
    float* logits = alloc((size_t)NGRAPH * NCLS + 8);
    float* meanb  = alloc(512);
    float* rstdb  = alloc(512);
    __bf16* abf   = (__bf16*)alloc((size_t)N_NODES * HDF / 2);  // bf16 A scratch
    __bf16* bbf   = (__bf16*)alloc(65536 / 2);                  // bf16 packed-B scratch
    (void)ws_size; (void)n_in; (void)in_sizes; (void)out_size;

    const int T = 256;
    auto gemm = [&](const float* A, const float* B, float* C, int M, int Nn, int K) {
        pack_a_kernel<<<cdiv((long long)M * K, T), T, 0, stream>>>(A, abf, (long long)M * K);
        int pbt = (K / 32) * (Nn / 16) * 32;
        pack_b_kernel<<<cdiv(pbt, T), T, 0, stream>>>(B, bbf, K, Nn);
        if (Nn % 64 == 0) {
            int strips = (M / 16) * (Nn / 64);
            gemm_wmma_bf16<4><<<cdiv(strips, 8), 256, 0, stream>>>(abf, bbf, C, M, Nn, K);
        } else {
            int strips = (M / 16) * (Nn / 32);
            gemm_wmma_bf16<2><<<cdiv(strips, 8), 256, 0, stream>>>(abf, bbf, C, M, Nn, K);
        }
    };
    auto bn = [&](float* xp, int rows, int C, const float* g, const float* b, int act) {
        bn_stats_kernel<<<C, 256, 0, stream>>>(xp, rows, C, meanb, rstdb);
        bn_apply_kernel<<<cdiv((long long)rows * C, T), T, 0, stream>>>(
            xp, meanb, rstdb, g, b, rows, C, act);
    };
    auto gat = [&](const float* hin, int Kin, const float* W, const float* a_s,
                   const float* a_d, const float* bias, float* out) {
        gemm(hin, W, tmp, N_NODES, HDF, Kin);
        att_score_kernel<<<cdiv(N_NODES * HEADS, T), T, 0, stream>>>(tmp, a_s, a_d, asrc, adst, N_NODES);
        att_init_self<<<cdiv(N_NODES * HEADS, T), T, 0, stream>>>(asrc, adst, emax, N_NODES);
        att_max_edges<<<cdiv(N_EDGES * HEADS, T), T, 0, stream>>>(asrc, adst, rowi, coli, emax, N_EDGES);
        att_den_init<<<cdiv(N_NODES * HEADS, T), T, 0, stream>>>(asrc, adst, emax, den, N_NODES);
        att_den_edges<<<cdiv(N_EDGES * HEADS, T), T, 0, stream>>>(asrc, adst, rowi, coli, emax, den, N_EDGES);
        att_alpha_edges<<<cdiv(N_EDGES * HEADS, T), T, 0, stream>>>(asrc, adst, rowi, coli, emax, den, ealpha, N_EDGES);
        att_out_init<<<cdiv((long long)N_NODES * HDF, T), T, 0, stream>>>(tmp, asrc, adst, emax, den, bias, out, N_NODES);
        att_scatter<<<cdiv((long long)N_EDGES * (HDF / 4), T), T, 0, stream>>>(tmp, rowi, coli, ealpha, out, N_EDGES);
    };

    // ---- encoder: GCN init + ReLU + BN ----
    fill_kernel<<<cdiv(N_NODES, T), T, 0, stream>>>(deg, 1.f, N_NODES);
    deg_kernel<<<cdiv(N_EDGES, T), T, 0, stream>>>(coli, nullptr, 0, 0, deg, N_EDGES);
    rsqrt_kernel<<<cdiv(N_NODES, T), T, 0, stream>>>(deg, N_NODES);
    hipMemcpyAsync(dinv, deg, N_NODES * sizeof(float), hipMemcpyDeviceToDevice, stream);
    gemm(x, W_init, tmp, N_NODES, HIDF, FIN);
    gcn_self_node<<<cdiv(N_NODES * HIDF, T), T, 0, stream>>>(tmp, dinv, b_init, h0, N_NODES, HIDF);
    gcn_scatter_kernel<<<cdiv((long long)N_EDGES * (HIDF / 4), T), T, 0, stream>>>(
        tmp, rowi, coli, dinv, nullptr, 0, 0, nullptr, h0, N_EDGES, HIDF);
    act_kernel<<<cdiv(N_NODES * HIDF, T), T, 0, stream>>>(h0, N_NODES * HIDF, 1);
    bn(h0, N_NODES, HIDF, gam0, bet0, 0);

    // ---- GAT 1 + BN + ELU ----
    gat(h0, HIDF, W1, as1, ad1, b1, hA);
    bn(hA, N_NODES, HDF, gam1, bet1, 2);

    // ---- GAT 2 + BN + ELU ----
    gat(hA, HDF, W2, as2, ad2, b2, hB);
    bn(hB, N_NODES, HDF, gam2, bet2, 2);

    // ---- edge attention scores (fused gather+linear+softmax2) ----
    edge_score_kernel<<<cdiv(N_EDGES, 8), 256, 0, stream>>>(hB, rowi, coli, Wea, bea, escore, N_EDGES);

    // ---- pooled graph counts ----
    fill_kernel<<<cdiv(NGRAPH, T), T, 0, stream>>>(cnt, 0.f, NGRAPH);
    pool_count<<<cdiv(N_NODES, T), T, 0, stream>>>(batch, cnt, N_NODES);

    // ---- causal / trivial weighted GCNs, fused directly into pooled sums ----
    for (int k = 0; k < 2; ++k) {
        const float* Wg = k == 0 ? Wgc : Wgt;
        const float* bg = k == 0 ? bgc : bgt;
        float* pool = k == 0 ? pc : pt;
        fill_kernel<<<cdiv(N_NODES, T), T, 0, stream>>>(deg, 1.f, N_NODES);
        deg_kernel<<<cdiv(N_EDGES, T), T, 0, stream>>>(coli, escore, 2, k, deg, N_EDGES);
        rsqrt_kernel<<<cdiv(N_NODES, T), T, 0, stream>>>(deg, N_NODES);
        hipMemcpyAsync(dinv, deg, N_NODES * sizeof(float), hipMemcpyDeviceToDevice, stream);
        gemm(hB, Wg, tmp, N_NODES, HDF, HDF);
        fill_kernel<<<cdiv(NGRAPH * HDF, T), T, 0, stream>>>(pool, 0.f, NGRAPH * HDF);
        gcn_self_pool<<<cdiv((long long)N_NODES * (HDF / 4), T), T, 0, stream>>>(
            tmp, dinv, batch, pool, N_NODES, HDF);
        gcn_scatter_kernel<<<cdiv((long long)N_EDGES * (HDF / 4), T), T, 0, stream>>>(
            tmp, rowi, coli, dinv, escore, 2, k, batch, pool, N_EDGES, HDF);
        pool_finalize<<<cdiv(NGRAPH * HDF, T), T, 0, stream>>>(pool, cnt, bg, NGRAPH, HDF);
    }

    concat_kernel<<<cdiv(NGRAPH * 2 * HDF, T), T, 0, stream>>>(pc, pt, comb, NGRAPH);

    // ---- three classifier heads -> log_softmax -> d_out ----
    const float* heads_in[3] = { pc, pt, comb };
    const int    heads_K [3] = { HDF, HDF, 2 * HDF };
    float* out = (float*)d_out;
    for (int k = 0; k < 3; ++k) {
        gemm(heads_in[k], clsW1[k], clsh, NGRAPH, HIDF, heads_K[k]);
        add_bias_kernel<<<cdiv(NGRAPH * HIDF, T), T, 0, stream>>>(clsh, clsB1[k], NGRAPH, HIDF);
        bn(clsh, NGRAPH, HIDF, clsG[k], clsBt[k], 1);
        final_linear_kernel<<<cdiv(NGRAPH * NCLS, T), T, 0, stream>>>(clsh, clsW2[k], clsB2[k], logits, NGRAPH);
        log_softmax_kernel<<<cdiv(NGRAPH, T), T, 0, stream>>>(logits, out + (size_t)k * NGRAPH * NCLS, NGRAPH);
    }
}
